// hypercorre_topk2_82008105550154
// MI455X (gfx1250) — matrix operand order repaired and checked by reading, not compiled, not run on hardware
//
#include <hip/hip_runtime.h>
#include <hip/hip_bf16.h>

// Problem constants (match reference)
#define BB 4
#define TT 4
#define CC 320
#define HWN 1600          // 40*40
#define THRESHV 0.95f
#define NEGV   (-1e9f)
#define EPSV   (1e-12f)

#define QTILES (HWN / 16)   // 100 query tiles per (b,t)
#define KPAIRS (HWN / 32)   // 50 key pairs (32 keys) per (b,t)
#define CCH    (CC / 32)    // 10 channel chunks of 32 for QK^T
#define CTL    (CC / 16)    // 20 channel tiles of 16 for PV

typedef _Float16 v8h  __attribute__((ext_vector_type(8)));
typedef _Float16 v16h __attribute__((ext_vector_type(16)));
typedef float    v8f  __attribute__((ext_vector_type(8)));

__device__ __forceinline__ v8f wmma_f16(v16h a, v16h b, v8f c) {
  return __builtin_amdgcn_wmma_f32_16x16x32_f16(
      /*neg_a=*/false, a, /*neg_b=*/false, b,
      /*c_mod=*/(short)0, c, /*reuse_a=*/false, /*reuse_b=*/false);
}

// B-matrix (32x16 f16) fragment: elems 0..7 = K = half8..+7, elems 8..15 = K = 16+half8..+7 (p already offset by half8)
__device__ __forceinline__ v16h fragB(const _Float16* p) {
  v8h lo = *(const v8h*)(p);
  v8h hi = *(const v8h*)(p + 16);
  return __builtin_shufflevector(lo, hi, 0,1,2,3,4,5,6,7,8,9,10,11,12,13,14,15);
}

// A-matrix (16x32 f16) fragment from an LDS row:
//   elems 0..7  = K = k0 + half8 .. +7
//   elems 8..15 = K = k0 + 16 + half8 .. +7
__device__ __forceinline__ v16h fragA(const _Float16* rowp, int k0, int half8) {
  v8h lo = *(const v8h*)(rowp + k0 + half8);
  v8h hi = *(const v8h*)(rowp + k0 + 16 + half8);
  return __builtin_shufflevector(lo, hi, 0,1,2,3,4,5,6,7,8,9,10,11,12,13,14,15);
}

// ---------------------------------------------------------------------------
// Kernel 1: inverse row norms for q (per b) and k (per b,t); zero qk-mask words
// ---------------------------------------------------------------------------
__global__ void sda_norms_kernel(const float* __restrict__ q,
                                 const float* __restrict__ k,
                                 float* __restrict__ invq,
                                 float* __restrict__ invk,
                                 unsigned* __restrict__ qkw) {
  int idx = blockIdx.x * blockDim.x + threadIdx.x;
  if (idx < BB * TT * KPAIRS) qkw[idx] = 0u;
  if (idx < BB * HWN) {
    int b = idx / HWN, p = idx % HWN;
    const float* base = q + (size_t)b * CC * HWN + p;
    float s = 0.f;
    for (int c = 0; c < CC; ++c) { float x = base[(size_t)c * HWN]; s += x * x; }
    invq[idx] = 1.0f / fmaxf(sqrtf(s), EPSV);
  }
  if (idx < BB * TT * HWN) {
    int bt = idx / HWN, p = idx % HWN;
    const float* base = k + (size_t)bt * CC * HWN + p;
    float s = 0.f;
    for (int c = 0; c < CC; ++c) { float x = base[(size_t)c * HWN]; s += x * x; }
    invk[idx] = 1.0f / fmaxf(sqrtf(s), EPSV);
  }
}

// ---------------------------------------------------------------------------
// Kernel 2/3: flash attention over keys for one (b, t, 16-query tile).
// SPARSE=false: cosine-sim logits, writes out_dense everywhere, builds masks.
// SPARSE=true : raw logits with -1e9 on matched keys, overwrites only rows
//               whose mask_kq is false.
// One wave32 per block. S^T = K*Q^T so exp(S) lands directly in PV A-fragment
// layout (lane L holds query L along the key axis).
// ---------------------------------------------------------------------------
template <bool SPARSE>
__global__ __launch_bounds__(32)
void sda_attn_kernel(const float* __restrict__ q, const float* __restrict__ k,
                     const float* __restrict__ v,
                     const float* __restrict__ invq,
                     const float* __restrict__ invk,
                     unsigned* __restrict__ kqm, unsigned* __restrict__ qkw,
                     float* __restrict__ out) {
  __shared__ __attribute__((aligned(16))) _Float16 qlds[16 * CC];   // 10 KB
  __shared__ __attribute__((aligned(16))) _Float16 klds[32 * CC];   // 20 KB
  __shared__ __attribute__((aligned(16))) _Float16 vtlds[CC * 32];  // 20 KB (transposed)

  const int lane  = threadIdx.x;
  const int li    = lane & 15;
  const int half8 = (lane & 16) ? 8 : 0;
  const int blk   = blockIdx.x;
  const int bt    = blk / QTILES;
  const int qt    = blk % QTILES;
  const int b     = bt / TT;
  const int q0    = qt * 16;
  const size_t kvbase = (size_t)bt * CC * HWN;

  // ---- stage Q tile (16 rows x 320 ch) as f16, row-major in channels ----
  {
    const float* qb = q + (size_t)b * CC * HWN + q0;
    for (int i = lane; i < 16 * CC; i += 32) {
      int row = i & 15, c = i >> 4;
      qlds[row * CC + c] = (_Float16)qb[(size_t)c * HWN + row];
    }
  }
  __syncthreads();

  const float iq = SPARSE ? 1.0f : invq[b * HWN + q0 + li];

  v8f acc[CTL];
#pragma unroll
  for (int ct = 0; ct < CTL; ++ct) acc[ct] = {};
  float m = -3e38f, l = 0.f;

  for (int pr = 0; pr < KPAIRS; ++pr) {
    const int j0 = pr * 32;
    // ---- stage K (row-major) and V^T (channel-major) for 32 keys ----
    {
      const float* kb = k + kvbase + j0;
      const float* vb = v + kvbase + j0;
      for (int c = 0; c < CC; ++c) {
        klds[lane * CC + c]  = (_Float16)kb[(size_t)c * HWN + lane];
        vtlds[c * 32 + lane] = (_Float16)vb[(size_t)c * HWN + lane];
      }
    }
    __syncthreads();

    // ---- S^T = K * Q^T : two 16(key)x16(query) tiles ----
    v8f s0 = {}, s1 = {};
    const _Float16* qrow  = qlds + li * CC;
    const _Float16* krow0 = klds + li * CC;
    const _Float16* krow1 = klds + (li + 16) * CC;
#pragma unroll
    for (int kk = 0; kk < CCH; ++kk) {
      const int k0 = kk * 32;
      v16h bq = fragB(qrow + k0 + half8);
      v16h a0 = fragA(krow0, k0, half8);
      v16h a1 = fragA(krow1, k0, half8);
      s0 = wmma_f16(a0, bq, s0);
      s1 = wmma_f16(a1, bq, s1);
    }

    if (!SPARSE) {
      // cosine scaling + hit masks. s0[r]: key j0+r+half8; s1[r]: key j0+16+r+half8
      float vk = invk[bt * HWN + j0 + lane];
      unsigned hitw = 0u;
#pragma unroll
      for (int r = 0; r < 8; ++r) {
        float ik0 = __shfl(vk, r + half8, 32);
        float ik1 = __shfl(vk, 16 + r + half8, 32);
        s0[r] *= iq * ik0;
        s1[r] *= iq * ik1;
        if (s0[r] >= THRESHV) hitw |= 1u << (r + half8);
        if (s1[r] >= THRESHV) hitw |= 1u << (16 + r + half8);
      }
      // OR over queries (lanes within each half), then merge the two halves
      hitw |= __shfl_xor(hitw, 1, 32);
      hitw |= __shfl_xor(hitw, 2, 32);
      hitw |= __shfl_xor(hitw, 4, 32);
      hitw |= __shfl_xor(hitw, 8, 32);
      hitw |= __shfl_xor(hitw, 16, 32);
      if (lane == 0 && hitw) atomicOr(&qkw[bt * KPAIRS + pr], hitw);
    } else {
      unsigned qw = qkw[bt * KPAIRS + pr];
#pragma unroll
      for (int r = 0; r < 8; ++r) {
        if ((qw >> (r + half8)) & 1u)        s0[r] = NEGV;
        if ((qw >> (16 + r + half8)) & 1u)   s1[r] = NEGV;
      }
    }

    // ---- online softmax (per query = per lane; halves synced via xor16) ----
    float lm = -3e38f;
#pragma unroll
    for (int r = 0; r < 8; ++r) lm = fmaxf(lm, fmaxf(s0[r], s1[r]));
    lm = fmaxf(lm, __shfl_xor(lm, 16, 32));
    float mn = fmaxf(m, lm);
    float sc = __expf(m - mn);
    float ps = 0.f;
    v8f p0, p1;
#pragma unroll
    for (int r = 0; r < 8; ++r) {
      p0[r] = __expf(s0[r] - mn);
      p1[r] = __expf(s1[r] - mn);
      ps += p0[r] + p1[r];
    }
    ps += __shfl_xor(ps, 16, 32);
    l = l * sc + ps;
    m = mn;

    // rescale accumulator: row M=r+half8 needs scale(query r+half8)
    float fr[8];
#pragma unroll
    for (int r = 0; r < 8; ++r) fr[r] = __shfl(sc, r + half8, 32);
#pragma unroll
    for (int ct = 0; ct < CTL; ++ct)
#pragma unroll
      for (int r = 0; r < 8; ++r) acc[ct][r] *= fr[r];

    // pack P into PV A-fragment (lane-local by construction of S^T layout)
    v16h pa;
#pragma unroll
    for (int e = 0; e < 8; ++e) { pa[e] = (_Float16)p0[e]; pa[e + 8] = (_Float16)p1[e]; }

    // ---- PV: acc[ct] += P(16x32) * V(32x16) per channel tile ----
#pragma unroll
    for (int ct = 0; ct < CTL; ++ct) {
      v16h vbf = fragB(vtlds + (ct * 16 + li) * 32 + half8);
      acc[ct] = wmma_f16(pa, vbf, acc[ct]);
    }
    __syncthreads();  // before restaging K/V next pair
  }

  // ---- epilogue ----
  float linv = 1.0f / l;
  float fr[8];
#pragma unroll
  for (int r = 0; r < 8; ++r) fr[r] = __shfl(linv, r + half8, 32);

  if (!SPARSE) {
#pragma unroll
    for (int ct = 0; ct < CTL; ++ct) {
      float* ob = out + kvbase + (size_t)(ct * 16 + li) * HWN + q0;
#pragma unroll
      for (int r = 0; r < 8; ++r) ob[r + half8] = acc[ct][r] * fr[r];
    }
    if (lane < 16) kqm[bt * HWN + q0 + lane] = (m >= THRESHV) ? 1u : 0u;
  } else {
    unsigned kql = kqm[bt * HWN + q0 + li];
    unsigned kqr[8];
#pragma unroll
    for (int r = 0; r < 8; ++r) kqr[r] = (unsigned)__shfl((int)kql, r + half8, 32);
#pragma unroll
    for (int ct = 0; ct < CTL; ++ct) {
      float* ob = out + kvbase + (size_t)(ct * 16 + li) * HWN + q0;
#pragma unroll
      for (int r = 0; r < 8; ++r)
        if (kqr[r] == 0u) ob[r + half8] = acc[ct][r] * fr[r];
    }
  }
}

extern "C" void kernel_launch(void* const* d_in, const int* in_sizes, int n_in,
                              void* d_out, int out_size, void* d_ws, size_t ws_size,
                              hipStream_t stream) {
  const float* q = (const float*)d_in[0];  // [B,1,C,H,W]
  const float* k = (const float*)d_in[1];  // [B,T,C,H,W]
  const float* v = (const float*)d_in[2];  // [B,T,C,H,W]
  float* out = (float*)d_out;              // [B,T,C,H,W]

  // workspace layout
  float*    invq = (float*)d_ws;                       // B*HW
  float*    invk = invq + BB * HWN;                    // B*T*HW
  unsigned* kqm  = (unsigned*)(invk + BB * TT * HWN);  // B*T*HW (mask_kq)
  unsigned* qkw  = kqm + BB * TT * HWN;                // B*T*KPAIRS bit words (mask_qk)

  const int nthreads = BB * TT * HWN;  // 25600 covers all three jobs in norms
  sda_norms_kernel<<<(nthreads + 255) / 256, 256, 0, stream>>>(q, k, invq, invk, qkw);

  dim3 grid(BB * TT * QTILES);  // 1600 single-wave workgroups
  sda_attn_kernel<false><<<grid, 32, 0, stream>>>(q, k, v, invq, invk, kqm, qkw, out);
  sda_attn_kernel<true ><<<grid, 32, 0, stream>>>(q, k, v, invq, invk, kqm, qkw, out);
}
